// MultiHeadedAttention_31799937859984
// MI455X (gfx1250) — matrix operand order repaired and checked by reading, not compiled, not run on hardware
//
#include <hip/hip_runtime.h>

// ---------------------------------------------------------------------------
// MHA forward (RoPE + causal flash attention) for MI455X / gfx1250.
// All matmuls on v_wmma_f32_16x16x32_bf16 (bf16 in, f32 accumulate).
// fp32 inputs/weights are converted to bf16 once up front so every GEMM
// k-loop is pure global_load_b128 + v_wmma (no conversion VALU in the loop).
// Attention computes S^T = K*Q^T so softmax is in-lane and the P fragment
// for P@V is produced directly in registers (no LDS round-trip).
// ---------------------------------------------------------------------------

typedef __attribute__((ext_vector_type(16))) __bf16        v16bf;
typedef __attribute__((ext_vector_type(2)))  __bf16        v2bf;
typedef __attribute__((ext_vector_type(8)))  float         v8f;
typedef __attribute__((ext_vector_type(4)))  float         f32x4;
typedef __attribute__((ext_vector_type(4)))  unsigned int  u32x4;

constexpr int Bc  = 4;
constexpr int Lc  = 2048;
constexpr int Dc  = 1024;
constexpr int Hc  = 16;
constexpr int hdc = 64;
constexpr int Mc  = Bc * Lc;   // 8192 rows
constexpr int Kc  = Dc;        // 1024 reduction dim

#define WMMA_BF16(A, B, C)                                                     \
  __builtin_amdgcn_wmma_f32_16x16x32_bf16(false, (A), false, (B), (short)0,    \
                                          (C), false, false)

union BFPack {
  v16bf v;
  unsigned int w[8];
  unsigned short u[16];
  u32x4 q[2];
};

__device__ __forceinline__ unsigned short f2bf(float f) {
  unsigned int u = __builtin_bit_cast(unsigned int, f);
  u += 0x7fffu + ((u >> 16) & 1u);  // RNE
  return (unsigned short)(u >> 16);
}

__device__ __forceinline__ unsigned int pack_bf16x2(float lo, float hi) {
#if __has_builtin(__builtin_amdgcn_cvt_pk_bf16_f32)
  v2bf t = __builtin_amdgcn_cvt_pk_bf16_f32(lo, hi);
  return __builtin_bit_cast(unsigned int, t);
#else
  unsigned int ulo = __builtin_bit_cast(unsigned int, lo);
  ulo += 0x7fffu + ((ulo >> 16) & 1u);
  unsigned int uhi = __builtin_bit_cast(unsigned int, hi);
  uhi += 0x7fffu + ((uhi >> 16) & 1u);
  return (ulo >> 16) | (uhi & 0xffff0000u);
#endif
}

// A-fragment (16x32 bf16, ISA 7.12.2): lane holds row (lane&15),
// K chunks {kb..kb+7, kb+16..kb+23}, kb = (lane>=16) ? 8 : 0.
__device__ __forceinline__ v16bf load_a_bf16(const unsigned short* __restrict__ A,
                                             int row, int lda, int kb0,
                                             int lane) {
  int kb = kb0 + ((lane & 16) ? 8 : 0);
  BFPack r;
  r.q[0] = *(const u32x4*)(A + (size_t)row * lda + kb);
  r.q[1] = *(const u32x4*)(A + (size_t)row * lda + kb + 16);
  return r.v;
}

// B-fragment (32x16 bf16): lane holds column (lane&15),
// K chunks {kb..kb+7, kb+16..kb+23}, kb = kb0 + ((lane>=16) ? 8 : 0).
__device__ __forceinline__ v16bf load_b_bf16(const unsigned short* __restrict__ Bp,
                                             int col, int ldb, int kb0,
                                             int lane) {
  int kb = kb0 + ((lane & 16) ? 8 : 0);
  const u32x4* p = (const u32x4*)(Bp + (size_t)col * ldb + kb);
  BFPack r;
  r.q[0] = p[0];
  r.q[1] = p[2];
  return r.v;
}

// ---------------------------------------------------------------------------
// One-shot fp32 -> bf16 conversion (bandwidth-bound, 8 elems/thread).
// ---------------------------------------------------------------------------
__global__ __launch_bounds__(256) void cvt_bf16_kernel(
    const float* __restrict__ in, unsigned short* __restrict__ out, int n8) {
  int i = blockIdx.x * 256 + threadIdx.x;
  if (i >= n8) return;
  const f32x4* p = (const f32x4*)(in + (size_t)i * 8);
  f32x4 a = p[0], b = p[1];
  u32x4 w;
  w[0] = pack_bf16x2(a[0], a[1]);
  w[1] = pack_bf16x2(a[2], a[3]);
  w[2] = pack_bf16x2(b[0], b[1]);
  w[3] = pack_bf16x2(b[2], b[3]);
  *(u32x4*)(out + (size_t)i * 8) = w;
}

// ---------------------------------------------------------------------------
// GEMM: Y = X @ W^T + b with X, W already bf16.  Wave computes a 64x32 tile
// (8 accumulators); inner loop is 12 global_load_b128 + 8 v_wmma per k-step.
// MODE 0: RoPE epilogue, out bf16 [B,H,L,hd]
// MODE 1: out bf16 transposed [B,H,hd,L]   (for V)
// MODE 2: out fp32 [M,N]                   (final projection)
// ---------------------------------------------------------------------------
template <int MODE>
__global__ __launch_bounds__(256) void gemm_wmma_kernel(
    const unsigned short* __restrict__ Ab, const unsigned short* __restrict__ Wb,
    const float* __restrict__ bias, void* __restrict__ Out) {
  const int lane = threadIdx.x & 31;
  const int wid  = threadIdx.x >> 5;
  constexpr int ntiles = Dc / 32;
  int task = blockIdx.x * 8 + wid;
  int mt = task / ntiles;
  int nt = task % ntiles;
  int m0 = mt * 64;
  int n0 = nt * 32;
  const int l15 = lane & 15;
  const int col0 = n0 + l15;
  const int col1 = col0 + 16;

  v8f z = {};
  v8f acc[4][2] = {{z, z}, {z, z}, {z, z}, {z, z}};

  for (int kb = 0; kb < Kc; kb += 32) {
    if (kb + 32 < Kc) {  // prefetch next k-slice (global_prefetch_b8)
      __builtin_prefetch(Wb + (size_t)col0 * Kc + kb + 32, 0, 1);
      __builtin_prefetch(Ab + (size_t)(m0 + l15) * Kc + kb + 32, 0, 1);
    }
    v16bf b0 = load_b_bf16(Wb, col0, Kc, kb, lane);
    v16bf b1 = load_b_bf16(Wb, col1, Kc, kb, lane);
#pragma unroll
    for (int t = 0; t < 4; ++t) {
      v16bf af = load_a_bf16(Ab, m0 + t * 16 + l15, Kc, kb, lane);
      acc[t][0] = WMMA_BF16(af, b0, acc[t][0]);
      acc[t][1] = WMMA_BF16(af, b1, acc[t][1]);
    }
  }

  const int b_ = m0 / Lc;  // Lc % 64 == 0 -> whole tile in one batch
  const int l0 = m0 % Lc;
  const int rboff = (lane & 16) ? 8 : 0;
  const float bv[2] = {bias[col0], bias[col1]};
#pragma unroll
  for (int c = 0; c < 2; ++c) {
    const int colW = c ? col1 : col0;
    const int h_ = colW >> 6;  // hd = 64
    const int d  = colW & 63;
    // inv_freq = 10000^(-2*(d/2)/hd) = exp(-(d/2)*ln(1e4)/32)
    const float invf = __expf(-(float)(d >> 1) * (9.2103403720f / 32.0f));
#pragma unroll
    for (int t = 0; t < 4; ++t) {
#pragma unroll
      for (int r = 0; r < 8; ++r) {
        const int l = l0 + t * 16 + r + rboff;
        float y = acc[t][c][r] + bv[c];
        if constexpr (MODE == 2) {
          ((float*)Out)[(size_t)(b_ * Lc + l) * Dc + colW] = y;
        } else if constexpr (MODE == 1) {
          ((unsigned short*)Out)[(((size_t)b_ * Hc + h_) * hdc + d) * Lc + l] =
              f2bf(y);
        } else {
          float ang = (float)l * invf;
          float cs = __cosf(ang), sn = __sinf(ang);
          float part = __shfl_xor(y, 1, 32);  // rope partner lane
          float out =
              (d & 1) ? fmaf(y, cs, part * sn) : fmaf(y, cs, -part * sn);
          ((unsigned short*)Out)[(((size_t)b_ * Hc + h_) * Lc + l) * hdc + d] =
              f2bf(out);
        }
      }
    }
  }
}

// ---------------------------------------------------------------------------
// Flash attention, S^T formulation. One wave owns 16 query rows.
//   S^T tile (keys x queries) via WMMA(A=K, B=Q^T): each lane then holds all
//   16 of its scores for one query in registers -> in-lane softmax, and the
//   resulting P values are already in the A-fragment layout for P@V.
// qh,kh: bf16 [B,H,L,hd]; vh: bf16 [B,H,hd,L]; attnb: bf16 [B,L,D].
// ---------------------------------------------------------------------------
__global__ __launch_bounds__(256) void attn_wmma_kernel(
    const unsigned short* __restrict__ qh, const unsigned short* __restrict__ kh,
    const unsigned short* __restrict__ vh, unsigned short* __restrict__ attnb) {
  __shared__ __align__(16) float Cbuf[8][16];
  const int lane = threadIdx.x & 31;
  const int wid  = threadIdx.x >> 5;
  float* cb = Cbuf[wid];

  constexpr int mtiles = Lc / 16;
  int task = blockIdx.x * 8 + wid;
  int mt = task % mtiles;
  int bh = task / mtiles;
  const int b_ = bh / Hc, h_ = bh % Hc;
  const int m0 = mt << 4;
  const int l15 = lane & 15;
  const int kbh = (lane & 16) ? 8 : 0;  // key sub-offset / O-row base

  const unsigned short* Q  = qh + (size_t)bh * Lc * hdc;
  const unsigned short* Kp = kh + (size_t)bh * Lc * hdc;
  const unsigned short* Vp = vh + (size_t)bh * hdc * Lc;

  // Q as B-fragments (constant across the key loop)
  v16bf qb0 = load_b_bf16(Q, m0 + l15, hdc, 0, lane);
  v16bf qb1 = load_b_bf16(Q, m0 + l15, hdc, 32, lane);

  const int qcol = m0 + l15;  // query tracked by this lane
  float mrun = -3.0e30f, lrun = 0.0f;
  v8f z = {};
  v8f o[4] = {z, z, z, z};
  const float scale = 0.125f;  // 1/sqrt(64)
  const int jend = (m0 + 15) >> 5;

  for (int j = 0; j <= jend; ++j) {
    const int t0 = j << 5;
    v8f st0 = z, st1 = z;
    {
      v16bf ka = load_a_bf16(Kp, t0 + l15, hdc, 0, lane);
      st0 = WMMA_BF16(ka, qb0, st0);
      ka = load_a_bf16(Kp, t0 + l15, hdc, 32, lane);
      st0 = WMMA_BF16(ka, qb1, st0);
      ka = load_a_bf16(Kp, t0 + 16 + l15, hdc, 0, lane);
      st1 = WMMA_BF16(ka, qb0, st1);
      ka = load_a_bf16(Kp, t0 + 16 + l15, hdc, 32, lane);
      st1 = WMMA_BF16(ka, qb1, st1);
    }

    // per-lane scores: p[0..7] = keys t0+kbh+r, p[8..15] = keys t0+16+kbh+r
    float p[16];
    const bool needMask = (t0 + 31) > m0;  // uniform: only diagonal tiles
#pragma unroll
    for (int r = 0; r < 8; ++r) {
      float a  = st0[r] * scale;
      float b2 = st1[r] * scale;
      if (needMask) {
        a  = (t0 + kbh + r      <= qcol) ? a  : -3.0e30f;
        b2 = (t0 + 16 + kbh + r <= qcol) ? b2 : -3.0e30f;
      }
      p[r]     = a;
      p[8 + r] = b2;
    }
    float mx = p[0];
#pragma unroll
    for (int i = 1; i < 16; ++i) mx = fmaxf(mx, p[i]);
    mx = fmaxf(mx, __shfl_xor(mx, 16, 32));  // sync lane pair (same query)
    const float mnew = fmaxf(mrun, mx);
    const float corr = __expf(mrun - mnew);
    mrun = mnew;
    float psum = 0.0f;
#pragma unroll
    for (int i = 0; i < 16; ++i) {
      p[i] = __expf(p[i] - mnew);
      psum += p[i];
    }
    lrun = lrun * corr + psum;  // per-lane partial (half the keys)

    // P is already in this lane's A-fragment layout: pack in registers.
    BFPack pk;
#pragma unroll
    for (int i = 0; i < 8; ++i) pk.w[i] = pack_bf16x2(p[2 * i], p[2 * i + 1]);

    // broadcast per-query corr into the O-row layout via per-wave LDS
    cb[l15] = corr;  // lane pair writes identical value
    asm volatile("s_wait_dscnt 0" ::: "memory");
    f32x4 c0 = *(const f32x4*)(cb + kbh);
    f32x4 c1 = *(const f32x4*)(cb + kbh + 4);
    v8f cv;
    cv[0] = c0[0]; cv[1] = c0[1]; cv[2] = c0[2]; cv[3] = c0[3];
    cv[4] = c1[0]; cv[5] = c1[1]; cv[6] = c1[2]; cv[7] = c1[3];
#pragma unroll
    for (int t = 0; t < 4; ++t) o[t] *= cv;

    // O += P @ V   (V transposed in memory: contiguous along t)
#pragma unroll
    for (int t = 0; t < 4; ++t) {
      v16bf vf = load_b_bf16(Vp, t * 16 + l15, Lc, t0, lane);
      o[t] = WMMA_BF16(pk.v, vf, o[t]);
    }
  }

  // combine lane-pair partials; broadcast 1/l into O-row layout
  lrun += __shfl_xor(lrun, 16, 32);
  cb[l15] = lrun;
  asm volatile("s_wait_dscnt 0" ::: "memory");
  f32x4 l0v = *(const f32x4*)(cb + kbh);
  f32x4 l1v = *(const f32x4*)(cb + kbh + 4);
  v8f lv;
  lv[0] = 1.0f / l0v[0]; lv[1] = 1.0f / l0v[1];
  lv[2] = 1.0f / l0v[2]; lv[3] = 1.0f / l0v[3];
  lv[4] = 1.0f / l1v[0]; lv[5] = 1.0f / l1v[1];
  lv[6] = 1.0f / l1v[2]; lv[7] = 1.0f / l1v[3];
#pragma unroll
  for (int t = 0; t < 4; ++t) {
    v8f val = o[t] * lv;
#pragma unroll
    for (int r = 0; r < 8; ++r) {
      const int row = m0 + r + kbh;
      const int d = t * 16 + l15;
      attnb[((size_t)b_ * Lc + row) * Dc + h_ * hdc + d] = f2bf(val[r]);
    }
  }
}

// ---------------------------------------------------------------------------
extern "C" void kernel_launch(void* const* d_in, const int* in_sizes, int n_in,
                              void* d_out, int out_size, void* d_ws,
                              size_t ws_size, hipStream_t stream) {
  (void)in_sizes; (void)n_in; (void)out_size; (void)ws_size;

  const float* q  = (const float*)d_in[0];
  const float* k  = (const float*)d_in[1];
  const float* v  = (const float*)d_in[2];
  const float* Wq = (const float*)d_in[3];
  const float* bq = (const float*)d_in[4];
  const float* Wk = (const float*)d_in[5];
  const float* bk = (const float*)d_in[6];
  const float* Wv = (const float*)d_in[7];
  const float* bv = (const float*)d_in[8];
  const float* Wo = (const float*)d_in[9];
  const float* bo = (const float*)d_in[10];

  const size_t plane = (size_t)Mc * Dc;  // 8M elements
  const size_t wsz   = (size_t)Dc * Dc;  // 1M elements
  unsigned short* p = (unsigned short*)d_ws;
  unsigned short* qb16 = p;  p += plane;
  unsigned short* kb16 = p;  p += plane;
  unsigned short* vb16 = p;  p += plane;
  unsigned short* wq16 = p;  p += wsz;
  unsigned short* wk16 = p;  p += wsz;
  unsigned short* wv16 = p;  p += wsz;
  unsigned short* wo16 = p;  p += wsz;
  unsigned short* qhb   = p; p += plane;
  unsigned short* khb   = p; p += plane;
  unsigned short* vhT   = p; p += plane;
  unsigned short* attnb = p;

  dim3 blk(256, 1, 1);
  const int inN8 = (int)(plane / 8);  // 1,048,576
  const int wN8  = (int)(wsz / 8);    // 131,072

  // one-shot fp32 -> bf16 conversions
  cvt_bf16_kernel<<<inN8 / 256, blk, 0, stream>>>(q, qb16, inN8);
  cvt_bf16_kernel<<<inN8 / 256, blk, 0, stream>>>(k, kb16, inN8);
  cvt_bf16_kernel<<<inN8 / 256, blk, 0, stream>>>(v, vb16, inN8);
  cvt_bf16_kernel<<<wN8 / 256, blk, 0, stream>>>(Wq, wq16, wN8);
  cvt_bf16_kernel<<<wN8 / 256, blk, 0, stream>>>(Wk, wk16, wN8);
  cvt_bf16_kernel<<<wN8 / 256, blk, 0, stream>>>(Wv, wv16, wN8);
  cvt_bf16_kernel<<<wN8 / 256, blk, 0, stream>>>(Wo, wo16, wN8);

  const int gemmGrid = (Mc / 64) * (Dc / 32) / 8;  // 512 blocks
  const int attnGrid = Bc * Hc * (Lc / 16) / 8;    // 1024 blocks

  gemm_wmma_kernel<0><<<gemmGrid, blk, 0, stream>>>(qb16, wq16, bq, qhb);
  gemm_wmma_kernel<0><<<gemmGrid, blk, 0, stream>>>(kb16, wk16, bk, khb);
  gemm_wmma_kernel<1><<<gemmGrid, blk, 0, stream>>>(vb16, wv16, bv, vhT);

  attn_wmma_kernel<<<attnGrid, blk, 0, stream>>>(qhb, khb, vhT, attnb);

  gemm_wmma_kernel<2><<<gemmGrid, blk, 0, stream>>>(attnb, wo16, bo, d_out);
}
